// LCNet_73521250173346
// MI455X (gfx1250) — compile-verified, matrix operand order checked by reading, not run on hardware
//
#include <hip/hip_runtime.h>
#include <hip/hip_bf16.h>

// ---------------------------------------------------------------------------
// 2-layer GCN for MI455X (gfx1250).
//
// Inputs (setup_inputs order):
//   d_in[0] x          [N,132] f32      (N = in_sizes[0]/132)
//   d_in[1] edge_index [2,E]   int64    (E = in_sizes[1]/2)
//   d_in[2] W1         [132,16] f32
//   d_in[3] b1         [16]    f32
//   d_in[4] W2         [16,10] f32
//   d_in[5] b2         [10]    f32
// Output: log_softmax(layer2) [N,10] f32.
//
// Design: edge gather/scatter dominates; all per-node tables are 16-float
// (64B) padded rows that live in the 192MB L2. Scatter uses native f32
// global atomics (clang::atomic metadata requests global_atomic_add_f32,
// not a CAS loop). The 132->16 transform uses v_wmma_f32_16x16x32_f16
// (one wave per 16-node tile, K padded 132->160). dinv[src] is folded into
// the stored rows; dinv[dst]+bias applied once per node after aggregation.
// ---------------------------------------------------------------------------

typedef __attribute__((ext_vector_type(16))) _Float16 v16h;
typedef __attribute__((ext_vector_type(8)))  float    v8f;

#define FDIM_IN  132
#define FPAD     16      // padded feature row (64B)
#define KCHUNKS  5       // ceil(132/32)

#ifndef __has_cpp_attribute
#define __has_cpp_attribute(x) 0
#endif

static __device__ __forceinline__ void atomic_add_f32(float* p, float v) {
#if __has_cpp_attribute(clang::atomic)
    // Request native global_atomic_add_f32 (no CAS expansion).
    [[clang::atomic(no_remote_memory, no_fine_grained_memory, ignore_denormal_mode)]] {
        __hip_atomic_fetch_add(p, v, __ATOMIC_RELAXED, __HIP_MEMORY_SCOPE_AGENT);
    }
#else
    __hip_atomic_fetch_add(p, v, __ATOMIC_RELAXED, __HIP_MEMORY_SCOPE_AGENT);
#endif
}

// ---- degree -----------------------------------------------------------------
__global__ void k_deg_init(float* __restrict__ deg, int N) {
    int i = blockIdx.x * blockDim.x + threadIdx.x;
    if (i < N) deg[i] = 1.0f;                      // self-loop contribution
}

__global__ void k_deg_edges(const long long* __restrict__ ei, float* __restrict__ deg,
                            long long E) {
    long long e = (long long)blockIdx.x * blockDim.x + threadIdx.x;
    if (e >= E) return;
    long long d = ei[E + e];                       // dst row
    atomic_add_f32(&deg[d], 1.0f);
}

__global__ void k_deg_to_dinv(float* __restrict__ deg, int N) {
    int i = blockIdx.x * blockDim.x + threadIdx.x;
    if (i < N) deg[i] = rsqrtf(deg[i]);            // deg >= 1 always
}

// ---- pack W1 into WMMA B fragments ------------------------------------------
// B fragment layout assumption (mirror of documented A 16-bit layout, ISA
// 7.12.2): lane l holds column n = l&15; with hi = l>>4, half j in [0,16):
//   k_local(j,hi) = (j<8) ? hi*8 + j : 16 + hi*8 + (j-8)
// One v16h per lane per K-chunk; K >= 132 zero-padded.
__global__ void k_pack_w1(const float* __restrict__ W1, _Float16* __restrict__ Bp) {
    int idx = blockIdx.x * blockDim.x + threadIdx.x;   // c*512 + lane*16 + j
    if (idx >= KCHUNKS * 32 * 16) return;
    int j    = idx & 15;
    int lane = (idx >> 4) & 31;
    int c    = idx >> 9;
    int n    = lane & 15;
    int hi   = lane >> 4;
    int kl   = (j < 8) ? (hi * 8 + j) : (16 + hi * 8 + (j - 8));
    int k    = c * 32 + kl;
    float v  = (k < FDIM_IN) ? W1[k * 16 + n] : 0.0f;
    Bp[idx]  = (_Float16)v;
}

// ---- layer-1 transform: hs = (x @ W1) * dinv, a1 = hs (self-loop init) ------
// One wave per 16-node tile. N is a multiple of 16 here (100000 = 6250*16);
// inactive waves exit whole (EXEC stays all-ones for WMMA).
__global__ void __launch_bounds__(128)
k_transform1_wmma(const float* __restrict__ x, const _Float16* __restrict__ Bp,
                  const float* __restrict__ dinv,
                  float* __restrict__ hs, float* __restrict__ a1, int N) {
    int wave = blockIdx.x * (blockDim.x >> 5) + (threadIdx.x >> 5);
    int lane = threadIdx.x & 31;
    int row0 = wave * 16;
    if (row0 >= N) return;

    int m  = lane & 15;
    int hi = lane >> 4;
    const float* xrow = x + (size_t)(row0 + m) * FDIM_IN;

    v8f acc = {};
    for (int c = 0; c < KCHUNKS; ++c) {
        v16h a;
        int kb = c * 32;
#pragma unroll
        for (int j = 0; j < 16; ++j) {
            int kl = (j < 8) ? (hi * 8 + j) : (16 + hi * 8 + (j - 8));
            int k  = kb + kl;
            float v = (k < FDIM_IN) ? xrow[k] : 0.0f;
            a[j] = (_Float16)v;
        }
        v16h b = *(const v16h*)(Bp + (c * 512 + lane * 16));
        acc = __builtin_amdgcn_wmma_f32_16x16x32_f16(
            /*neg_a=*/false, a, /*neg_b=*/false, b,
            /*c_mod=*/(short)0, acc, /*reuse_a=*/false, /*reuse_b=*/false);
    }

    // C layout: VGPR r -> row (hi*8 + r), col = lane&15. Writes per r are
    // 16 contiguous floats (64B) -> coalesced.
    int n = lane & 15;
#pragma unroll
    for (int r = 0; r < 8; ++r) {
        int row = row0 + hi * 8 + r;
        float v = acc[r] * dinv[row];
        hs[(size_t)row * FPAD + n] = v;
        a1[(size_t)row * FPAD + n] = v;
    }
}

// ---- edge aggregation: acc[dst,f] += hs[src,f] ------------------------------
// 16 consecutive lanes handle one edge: lane 0 of each group loads the two
// int64 endpoints once and broadcasts via shfl; the gather is one contiguous
// 64B row per group; one native f32 atomic per active lane (F=16 or 10).
__global__ void k_agg_edges(const long long* __restrict__ ei,
                            const float* __restrict__ hsrc,
                            float* __restrict__ acc, long long E, int F) {
    long long t = (long long)blockIdx.x * blockDim.x + threadIdx.x;
    long long e = t >> 4;              // group-uniform
    int f = (int)(t & 15);
    if (e >= E) return;                // group-uniform exit

    long long s = 0, d = 0;
    if (f == 0) {
        s = ei[e];
        d = ei[E + e];
    }
    s = __shfl(s, 0, 16);
    d = __shfl(d, 0, 16);

    if (f >= F) return;
    float v = hsrc[s * FPAD + f];
    atomic_add_f32(&acc[d * FPAD + f], v);
}

// ---- layer-2 transform (fused): z1 = relu(dinv*a1 + b1);
//      hs2 = (z1 @ W2) * dinv (padded to 16 cols); a2 = hs2 (self-loop init) --
__global__ void k_layer2(const float* __restrict__ a1, const float* __restrict__ dinv,
                         const float* __restrict__ W2, const float* __restrict__ b1,
                         float* __restrict__ hs2, float* __restrict__ a2, int N) {
    __shared__ float sW2[16 * 10];
    __shared__ float sb1[16];
    if (threadIdx.x < 160) sW2[threadIdx.x] = W2[threadIdx.x];
    if (threadIdx.x < 16)  sb1[threadIdx.x] = b1[threadIdx.x];
    __syncthreads();

    int i = blockIdx.x * blockDim.x + threadIdx.x;
    if (i >= N) return;
    float dv = dinv[i];
    const float* ar = a1 + (size_t)i * FPAD;

    float z[16];
#pragma unroll
    for (int f = 0; f < 16; ++f) {
        float t = dv * ar[f] + sb1[f];
        z[f] = t > 0.0f ? t : 0.0f;
    }
    float o[10];
#pragma unroll
    for (int j = 0; j < 10; ++j) o[j] = 0.0f;
#pragma unroll
    for (int f = 0; f < 16; ++f) {
        float zf = z[f];
#pragma unroll
        for (int j = 0; j < 10; ++j) o[j] = fmaf(zf, sW2[f * 10 + j], o[j]);
    }
    float* hr = hs2 + (size_t)i * FPAD;
    float* aw = a2  + (size_t)i * FPAD;
#pragma unroll
    for (int j = 0; j < 10; ++j) { float v = o[j] * dv; hr[j] = v; aw[j] = v; }
#pragma unroll
    for (int j = 10; j < 16; ++j) { hr[j] = 0.0f; aw[j] = 0.0f; }
}

// ---- finalize: z2 = dinv*a2 + b2; log_softmax over 10 -----------------------
__global__ void k_finalize(const float* __restrict__ a2, const float* __restrict__ dinv,
                           const float* __restrict__ b2, float* __restrict__ out, int N) {
    __shared__ float sb2[10];
    if (threadIdx.x < 10) sb2[threadIdx.x] = b2[threadIdx.x];
    __syncthreads();

    int i = blockIdx.x * blockDim.x + threadIdx.x;
    if (i >= N) return;
    float dv = dinv[i];
    const float* ar = a2 + (size_t)i * FPAD;

    float z[10];
    float mx = -1e30f;
#pragma unroll
    for (int j = 0; j < 10; ++j) {
        z[j] = dv * ar[j] + sb2[j];
        mx = fmaxf(mx, z[j]);
    }
    float s = 0.0f;
#pragma unroll
    for (int j = 0; j < 10; ++j) s += __expf(z[j] - mx);
    float ls = __logf(s);
#pragma unroll
    for (int j = 0; j < 10; ++j) out[(size_t)i * 10 + j] = z[j] - mx - ls;
}

// ---------------------------------------------------------------------------
extern "C" void kernel_launch(void* const* d_in, const int* in_sizes, int n_in,
                              void* d_out, int out_size, void* d_ws, size_t ws_size,
                              hipStream_t stream) {
    const float*     x  = (const float*)d_in[0];
    const long long* ei = (const long long*)d_in[1];
    const float*     W1 = (const float*)d_in[2];
    const float*     b1 = (const float*)d_in[3];
    const float*     W2 = (const float*)d_in[4];
    const float*     b2 = (const float*)d_in[5];
    float* out = (float*)d_out;

    const int       N = in_sizes[0] / FDIM_IN;
    const long long E = (long long)in_sizes[1] / 2;

    // Workspace carve (all offsets 256B aligned; ~13.3 MB total for N=100k):
    //   Bpack  : 5*32*16 f16 fragments of W1
    //   dinv   : [N] f32 (deg in-place -> rsqrt)
    //   buf_h  : [N,16] f32  (hs layer1, then hs2 layer2)
    //   buf_a  : [N,16] f32  (a1 layer1, then a2 layer2)
    char* ws = (char*)d_ws;
    size_t off = 0;
    _Float16* Bpack = (_Float16*)(ws + off); off += ((size_t)KCHUNKS * 512 * 2 + 255) & ~(size_t)255;
    float* dinv  = (float*)(ws + off); off += ((size_t)N * 4 + 255) & ~(size_t)255;
    float* buf_h = (float*)(ws + off); off += ((size_t)N * FPAD * 4 + 255) & ~(size_t)255;
    float* buf_a = (float*)(ws + off); off += ((size_t)N * FPAD * 4 + 255) & ~(size_t)255;
    (void)ws_size; (void)n_in; (void)out_size;

    const int B = 256;
    const int gN = (N + B - 1) / B;

    // 1) degree (with self-loops) -> dinv
    k_deg_init<<<gN, B, 0, stream>>>(dinv, N);
    {
        long long t = E;
        int g = (int)((t + B - 1) / B);
        k_deg_edges<<<g, B, 0, stream>>>(ei, dinv, E);
    }
    k_deg_to_dinv<<<gN, B, 0, stream>>>(dinv, N);

    // 2) pack W1 fragments, then WMMA transform + self-loop init
    k_pack_w1<<<(KCHUNKS * 512 + B - 1) / B, B, 0, stream>>>(W1, Bpack);
    {
        int tiles = (N + 15) / 16;          // waves
        int wpb = 4;                         // 128-thread blocks, 4 waves
        int g = (tiles + wpb - 1) / wpb;
        k_transform1_wmma<<<g, 32 * wpb, 0, stream>>>(x, Bpack, dinv, buf_h, buf_a, N);
    }

    // 3) layer-1 edge aggregation (16 lanes / edge)
    {
        long long t = E * 16;
        int g = (int)((t + B - 1) / B);
        k_agg_edges<<<g, B, 0, stream>>>(ei, buf_h, buf_a, E, 16);
    }

    // 4) fused relu/bias + 16x10 transform + self-loop init (overwrites buffers
    //    index-for-index after reading them; per-thread RAW is safe)
    k_layer2<<<gN, B, 0, stream>>>(buf_a, dinv, W2, b1, buf_h, buf_a, N);

    // 5) layer-2 edge aggregation (10 active lanes of 16)
    {
        long long t = E * 16;
        int g = (int)((t + B - 1) / B);
        k_agg_edges<<<g, B, 0, stream>>>(ei, buf_h, buf_a, E, 10);
    }

    // 6) dinv * acc + b2, log_softmax -> d_out
    k_finalize<<<gN, B, 0, stream>>>(buf_a, dinv, b2, out, N);
}
